// MultiHeadedAttention_48816598287017
// MI455X (gfx1250) — matrix operand order repaired and checked by reading, not compiled, not run on hardware
//
#include <hip/hip_runtime.h>
#include <hip/hip_bf16.h>
#include <stdint.h>

// Problem constants (match reference)
#define B_  4
#define S_  2048
#define D_  1024
#define H_  16
#define DK_ 64

#define LDT 72   // padded LDS row stride in bf16 elems: 144B = 16B-aligned, 36-bank stride

typedef __attribute__((ext_vector_type(16))) __bf16 v16bf;
typedef __attribute__((ext_vector_type(8)))  float  v8f;

__device__ __forceinline__ unsigned short f32_to_bf16(float x) {
  unsigned u = __float_as_uint(x);
  return (unsigned short)((u + 0x7FFFu + ((u >> 16) & 1u)) >> 16);  // RNE
}
__device__ __forceinline__ unsigned int pack2_bf16(float x, float y) {
  unsigned ux = __float_as_uint(x); ux = (ux + 0x7FFFu + ((ux >> 16) & 1u)) >> 16;
  unsigned uy = __float_as_uint(y); uy = (uy + 0x7FFFu + ((uy >> 16) & 1u)) >> 16;
  return (ux & 0xFFFFu) | (uy << 16);
}

union V16Cast { struct { uint4 lo, hi; } q; v16bf v; };

// A operand (16x32 bf16): lane = M (laneLo), elem j<8 -> K=32c+8*laneHi+j,
// elem j>=8 -> K=32c+16+8*laneHi+(j-8).  row = pointer to LDS row (stride LDT applied by caller).
__device__ __forceinline__ v16bf lds_load_a(const unsigned short* row, int c, int laneHi) {
  V16Cast u;
  u.q.lo = *(const uint4*)(row + c * 32 + laneHi * 8);
  u.q.hi = *(const uint4*)(row + c * 32 + 16 + laneHi * 8);
  return u.v;
}
// B operand (32x16 bf16): lane = N (laneLo), elem j<8 -> K=32c+8*laneHi+j, elem j>=8 -> K=32c+16+8*laneHi+(j-8)
__device__ __forceinline__ v16bf lds_load_b(const unsigned short* row, int c, int laneHi) {
  V16Cast u;
  const unsigned short* p = row + c * 32 + laneHi * 8;
  u.q.lo = *(const uint4*)(p);
  u.q.hi = *(const uint4*)(p + 16);
  return u.v;
}

// ---------------------------------------------------------------------------
// Projection GEMM:  Y[m,n] = sum_d X[m,d] * W[n,d] + bias[n]   (torch x@W.T+b)
// mode 0: bf16 out at [b,h,s,dk]   (Q, K)
// mode 1: bf16 out at [b,h,dk,s]   (V transposed, so attention PV B-operand is contiguous)
// mode 2: f32  out at [m, n]       (final output projection)
// Block: 256 threads (8 waves). Tile 128(M) x 64(N), K stepped by 64 via LDS.
// ---------------------------------------------------------------------------
__global__ __launch_bounds__(256) void proj_gemm_kernel(
    const float* __restrict__ X, const float* __restrict__ W,
    const float* __restrict__ bias, unsigned short* __restrict__ out_bf,
    float* __restrict__ out_f, int mode)
{
  __shared__ __align__(16) unsigned short Alds[128 * LDT];
  __shared__ __align__(16) unsigned short Blds[64 * LDT];

  const int tid    = threadIdx.x;
  const int wave   = tid >> 5;
  const int lane   = tid & 31;
  const int laneLo = lane & 15;
  const int laneHi = lane >> 4;
  const int m0 = blockIdx.y * 128;
  const int n0 = blockIdx.x * 64;
  const int K  = D_;

  v8f zero = {0.f,0.f,0.f,0.f,0.f,0.f,0.f,0.f};
  v8f acc[4];
#pragma unroll
  for (int i = 0; i < 4; ++i) acc[i] = zero;

  for (int k0 = 0; k0 < K; k0 += 64) {
    __syncthreads();
    // --- stage X tile [128 x 64] f32 -> bf16 LDS ---
    {
      int r  = tid >> 1;
      int cb = (tid & 1) * 32;
      const float4* src = (const float4*)(X + (size_t)(m0 + r) * K + k0 + cb);
      if (k0 + 64 < K) __builtin_prefetch(X + (size_t)(m0 + r) * K + k0 + 64 + cb, 0, 0);
      unsigned short* dst = Alds + r * LDT + cb;
#pragma unroll
      for (int j = 0; j < 4; ++j) {
        float4 f0 = src[2 * j], f1 = src[2 * j + 1];
        uint4 o;
        o.x = pack2_bf16(f0.x, f0.y); o.y = pack2_bf16(f0.z, f0.w);
        o.z = pack2_bf16(f1.x, f1.y); o.w = pack2_bf16(f1.z, f1.w);
        *(uint4*)(dst + 8 * j) = o;
      }
    }
    // --- stage W tile [64 x 64] f32 -> bf16 LDS (rows are output cols n) ---
    if (tid < 128) {
      int r  = tid >> 1;
      int cb = (tid & 1) * 32;
      const float4* src = (const float4*)(W + (size_t)(n0 + r) * K + k0 + cb);
      unsigned short* dst = Blds + r * LDT + cb;
#pragma unroll
      for (int j = 0; j < 4; ++j) {
        float4 f0 = src[2 * j], f1 = src[2 * j + 1];
        uint4 o;
        o.x = pack2_bf16(f0.x, f0.y); o.y = pack2_bf16(f0.z, f0.w);
        o.z = pack2_bf16(f1.x, f1.y); o.w = pack2_bf16(f1.z, f1.w);
        *(uint4*)(dst + 8 * j) = o;
      }
    }
    __syncthreads();

    // --- 8 WMMAs per wave per K-step ---
#pragma unroll
    for (int c = 0; c < 2; ++c) {
      v16bf a = lds_load_a(Alds + (wave * 16 + laneLo) * LDT, c, laneHi);
#pragma unroll
      for (int ni = 0; ni < 4; ++ni) {
        v16bf b = lds_load_b(Blds + (ni * 16 + laneLo) * LDT, c, laneHi);
        acc[ni] = __builtin_amdgcn_wmma_f32_16x16x32_bf16(
            false, a, false, b, (short)0, acc[ni], false, false);
      }
    }
  }

  // --- epilogue: bias + layout-specific store ---
#pragma unroll
  for (int ni = 0; ni < 4; ++ni) {
    int n = n0 + ni * 16 + laneLo;
    float bv = bias[n];
#pragma unroll
    for (int i = 0; i < 8; ++i) {
      int m = m0 + wave * 16 + i + 8 * laneHi;
      float v = acc[ni][i] + bv;
      if (mode == 2) {
        out_f[(size_t)m * D_ + n] = v;
      } else {
        int b = m >> 11, s = m & (S_ - 1);     // S_ = 2048
        int h = n >> 6, dk = n & (DK_ - 1);    // DK_ = 64
        size_t idx = (mode == 0)
            ? (((size_t)(b * H_ + h) * S_) + s) * DK_ + dk
            : (((size_t)(b * H_ + h) * DK_) + dk) * S_ + s;
        out_bf[idx] = f32_to_bf16(v);
      }
    }
  }
}

// ---------------------------------------------------------------------------
// Flash attention: one workgroup = 64 query rows of one (b,h). 128 thr = 4 waves.
// Online softmax with running (m,l); O accumulated in WMMA f32 regs.
// ---------------------------------------------------------------------------
__global__ __launch_bounds__(128) void flash_attn_kernel(
    const unsigned short* __restrict__ Qb,   // [B,H,S,DK] bf16
    const unsigned short* __restrict__ Kb,   // [B,H,S,DK] bf16
    const unsigned short* __restrict__ Vt,   // [B,H,DK,S] bf16
    const int* __restrict__ mask,            // [B,S,S] int32
    float* __restrict__ ctx)                 // [B,S,D] f32
{
  __shared__ __align__(16) unsigned short Qlds[64 * LDT];
  __shared__ __align__(16) unsigned short Klds[64 * LDT];   // rows = key pos, cols = d
  __shared__ __align__(16) unsigned short Vlds[64 * LDT];   // rows = d, cols = key pos
  __shared__ __align__(16) unsigned short Plds[64 * LDT];   // rows = q, cols = key pos
  __shared__ float Slds[64 * 68];
  __shared__ float m_s[64], l_s[64], c_s[64];

  const int tid    = threadIdx.x;
  const int wave   = tid >> 5;
  const int lane   = tid & 31;
  const int laneLo = lane & 15;
  const int laneHi = lane >> 4;
  const int q0 = blockIdx.x * 64;
  const int bh = blockIdx.y;
  const int b  = bh >> 4;
  const int h  = bh & 15;

  // stage Q tile once
  {
    int r = tid >> 1, cb = (tid & 1) * 32;
    const uint4* src = (const uint4*)(Qb + ((size_t)bh * S_ + q0 + r) * DK_ + cb);
    uint4* dst = (uint4*)(Qlds + r * LDT + cb);
#pragma unroll
    for (int j = 0; j < 4; ++j) dst[j] = src[j];
  }
  if (tid < 64) { m_s[tid] = -1e30f; l_s[tid] = 0.f; }

  v8f zero = {0.f,0.f,0.f,0.f,0.f,0.f,0.f,0.f};
  v8f oacc[4];
#pragma unroll
  for (int i = 0; i < 4; ++i) oacc[i] = zero;

  for (int k0 = 0; k0 < S_; k0 += 64) {
    __syncthreads();
    // stage K and Vt tiles
    {
      int r = tid >> 1, cb = (tid & 1) * 32;
      const uint4* sk = (const uint4*)(Kb + ((size_t)bh * S_ + k0 + r) * DK_ + cb);
      uint4* dk = (uint4*)(Klds + r * LDT + cb);
      const uint4* sv = (const uint4*)(Vt + ((size_t)bh * DK_ + r) * S_ + k0 + cb);
      uint4* dv = (uint4*)(Vlds + r * LDT + cb);
      if (k0 + 64 < S_) {
        __builtin_prefetch(Kb + ((size_t)bh * S_ + k0 + 64 + r) * DK_ + cb, 0, 0);
        __builtin_prefetch(Vt + ((size_t)bh * DK_ + r) * S_ + k0 + 64 + cb, 0, 0);
      }
#pragma unroll
      for (int j = 0; j < 4; ++j) { dk[j] = sk[j]; dv[j] = sv[j]; }
    }
    __syncthreads();

    // S = Q @ K^T  (K-dim = d, 2 chunks of 32)
    v8f sacc[4];
#pragma unroll
    for (int i = 0; i < 4; ++i) sacc[i] = zero;
#pragma unroll
    for (int c = 0; c < 2; ++c) {
      v16bf a = lds_load_a(Qlds + (wave * 16 + laneLo) * LDT, c, laneHi);
#pragma unroll
      for (int ni = 0; ni < 4; ++ni) {
        v16bf bm = lds_load_b(Klds + (ni * 16 + laneLo) * LDT, c, laneHi);
        sacc[ni] = __builtin_amdgcn_wmma_f32_16x16x32_bf16(
            false, a, false, bm, (short)0, sacc[ni], false, false);
      }
    }
    // scale, mask, spill to LDS
#pragma unroll
    for (int ni = 0; ni < 4; ++ni) {
#pragma unroll
      for (int i = 0; i < 8; ++i) {
        float sv = sacc[ni][i] * 0.125f;   // 1/sqrt(64)
        int qg = q0 + wave * 16 + i + 8 * laneHi;
        int kg = k0 + ni * 16 + laneLo;
        if (mask[((size_t)b * S_ + qg) * S_ + kg] == 0) sv = -1e9f;
        Slds[(wave * 16 + i + 8 * laneHi) * 68 + ni * 16 + laneLo] = sv;
      }
    }
    __syncthreads();

    // online softmax: one thread per row
    if (tid < 64) {
      const float* srow = Slds + tid * 68;
      float mo = m_s[tid];
      float mx = srow[0];
      for (int c = 1; c < 64; ++c) mx = fmaxf(mx, srow[c]);
      float mn = fmaxf(mo, mx);
      float corr = __expf(mo - mn);
      float sum = 0.f;
      unsigned short* prow = Plds + tid * LDT;
      for (int c = 0; c < 64; ++c) {
        float p = __expf(srow[c] - mn);
        sum += p;
        prow[c] = f32_to_bf16(p);
      }
      m_s[tid] = mn;
      l_s[tid] = l_s[tid] * corr + sum;
      c_s[tid] = corr;
    }
    __syncthreads();

    // rescale O, then O += P @ V  (K-dim = key pos)
    float cr[8];
#pragma unroll
    for (int i = 0; i < 8; ++i) cr[i] = c_s[wave * 16 + i + 8 * laneHi];
#pragma unroll
    for (int ti = 0; ti < 4; ++ti)
#pragma unroll
      for (int i = 0; i < 8; ++i) oacc[ti][i] *= cr[i];
#pragma unroll
    for (int c = 0; c < 2; ++c) {
      v16bf a = lds_load_a(Plds + (wave * 16 + laneLo) * LDT, c, laneHi);
#pragma unroll
      for (int ti = 0; ti < 4; ++ti) {
        v16bf bm = lds_load_b(Vlds + (ti * 16 + laneLo) * LDT, c, laneHi);
        oacc[ti] = __builtin_amdgcn_wmma_f32_16x16x32_bf16(
            false, a, false, bm, (short)0, oacc[ti], false, false);
      }
    }
  }

  // epilogue: normalize by l, write ctx[b, q, h*64 + d] (f32)
  float inv[8];
#pragma unroll
  for (int i = 0; i < 8; ++i) inv[i] = 1.0f / l_s[wave * 16 + i + 8 * laneHi];
#pragma unroll
  for (int ti = 0; ti < 4; ++ti) {
#pragma unroll
    for (int i = 0; i < 8; ++i) {
      int qg = q0 + wave * 16 + i + 8 * laneHi;
      int d  = ti * 16 + laneLo;
      ctx[((size_t)b * S_ + qg) * D_ + h * DK_ + d] = oacc[ti][i] * inv[i];
    }
  }
}

// ---------------------------------------------------------------------------
extern "C" void kernel_launch(void* const* d_in, const int* in_sizes, int n_in,
                              void* d_out, int out_size, void* d_ws, size_t ws_size,
                              hipStream_t stream) {
  const float* query = (const float*)d_in[0];
  const float* key   = (const float*)d_in[1];
  const float* value = (const float*)d_in[2];
  const int*   mask  = (const int*)  d_in[3];
  const float* Wq = (const float*)d_in[4];   const float* bq = (const float*)d_in[5];
  const float* Wk = (const float*)d_in[6];   const float* bk = (const float*)d_in[7];
  const float* Wv = (const float*)d_in[8];   const float* bv = (const float*)d_in[9];
  const float* Wo = (const float*)d_in[10];  const float* bo = (const float*)d_in[11];
  float* out = (float*)d_out;

  // workspace: Qb, Kb (bf16 [B,H,S,DK]), Vt (bf16 [B,H,DK,S]), ctx (f32 [B,S,D])
  char* ws = (char*)d_ws;
  const size_t bhsd = (size_t)B_ * H_ * S_ * DK_;   // 8,388,608 elems
  unsigned short* Qb = (unsigned short*)ws;  ws += bhsd * sizeof(unsigned short);
  unsigned short* Kb = (unsigned short*)ws;  ws += bhsd * sizeof(unsigned short);
  unsigned short* Vt = (unsigned short*)ws;  ws += bhsd * sizeof(unsigned short);
  float* ctx = (float*)ws;                           // 33.5 MB

  dim3 gg(D_ / 64, (B_ * S_) / 128);   // (16, 64)
  proj_gemm_kernel<<<gg, 256, 0, stream>>>(query, Wq, bq, Qb, nullptr, 0);
  proj_gemm_kernel<<<gg, 256, 0, stream>>>(key,   Wk, bk, Kb, nullptr, 0);
  proj_gemm_kernel<<<gg, 256, 0, stream>>>(value, Wv, bv, Vt, nullptr, 1);

  dim3 ga(S_ / 64, B_ * H_);           // (32, 64)
  flash_attn_kernel<<<ga, 128, 0, stream>>>(Qb, Kb, Vt, mask, ctx);

  proj_gemm_kernel<<<gg, 256, 0, stream>>>(ctx, Wo, bo, nullptr, out, 2);
}
